// CrossAttention_53755810677135
// MI455X (gfx1250) — matrix operand order repaired and checked by reading, not compiled, not run on hardware
//
#include <hip/hip_runtime.h>
#include <math.h>

#define HWD   16384
#define HH    128
#define WW    128
#define BB    2
#define DIMC  384
#define HEADS 8
#define CHD   48          // DIMC / HEADS
#define QKVC  1152        // 3 * DIMC

typedef __bf16 bf16_t;
typedef __attribute__((ext_vector_type(16))) __bf16 v16bf;
typedef __attribute__((ext_vector_type(8)))  float  v8f;
typedef __attribute__((ext_vector_type(4)))  unsigned int u32x4;
typedef __attribute__((ext_vector_type(8)))  unsigned int u32x8;

__device__ __forceinline__ bf16_t tobf(float f) { return (bf16_t)f; }

// A-matrix (16x32, 16-bit) element->K mapping per ISA 7.12.2
__device__ __forceinline__ int ka_idx(int e, int half) {
    return e + 8 * half + (e >= 8 ? 8 : 0);
}
// B-matrix (32x16, 16-bit) element->K mapping (same K order as A)
__device__ __forceinline__ int kb_idx(int e, int half) {
    return e + 8 * half + (e >= 8 ? 8 : 0);
}

// Low 32 bits of a generic LDS pointer == LDS byte offset (flat aperture rule,
// ISA 10.2: only the low 32 address bits are used).
__device__ __forceinline__ unsigned lds_off(const void* p) {
    return (unsigned)(unsigned long long)(uintptr_t)p;
}

// ---------------------------------------------------------------------------
// Tensor Data Mover: 2-D tile load global->LDS (bf16 elements).
// D# built per CDNA5 ISA (2 SGPR groups, 2-D).
//   tile_d0 = contiguous dim length (elements), tile_d1 = #rows,
//   stride0 = row stride in elements.  Tracked by TENSORcnt.
// ---------------------------------------------------------------------------
__device__ __forceinline__ void tdm_load_2d(unsigned lds_addr, const void* gptr,
                                            unsigned tile_d0, unsigned tile_d1,
                                            unsigned long long stride0)
{
    unsigned long long ga = (unsigned long long)(uintptr_t)gptr;
    const unsigned td0 = 1u << 24, td1 = 1u << 24;   // huge tensor dims: interior tile, no OOB
    u32x4 g0;
    g0[0] = 1u;                                       // count=1, is_restore=0, gather off
    g0[1] = lds_addr;                                 // LDS byte address
    g0[2] = (unsigned)ga;                             // global addr [31:0]
    g0[3] = ((unsigned)(ga >> 32) & 0x01ffffffu) | (2u << 30);  // addr[56:32] | type=2
    u32x8 g1;
    g1[0] = 1u << 16;                                 // data_size=1 (2 bytes); mask=0; no flags
    g1[1] = (td0 & 0xffffu) << 16;                    // [63:48]  tensor_dim0 lo16
    g1[2] = (td0 >> 16) | ((td1 & 0xffffu) << 16);    // dim0 hi16 | dim1 lo16
    g1[3] = (td1 >> 16) | (tile_d0 << 16);            // dim1 hi16 | tile_dim0
    g1[4] = tile_d1;                                  // tile_dim1 | tile_dim2=0
    g1[5] = (unsigned)(stride0 & 0xffffffffu);        // tensor_dim0_stride lo32
    g1[6] = (unsigned)(stride0 >> 32) & 0xffffu;      // stride hi16 | dim1_stride lo16 = 0
    g1[7] = 0u;
    asm volatile("tensor_load_to_lds %0, %1" :: "s"(g0), "s"(g1) : "memory");
}

// Per-lane async copy global->LDS, 16 bytes, tracked by ASYNCcnt.
__device__ __forceinline__ void async_ld_b128(unsigned lds_addr, const void* g) {
    asm volatile("global_load_async_to_lds_b128 %0, %1, off"
                 :: "v"(lds_addr), "v"((unsigned long long)(uintptr_t)g) : "memory");
}
__device__ __forceinline__ void wait_async0() {
    asm volatile("s_wait_asynccnt 0x0" ::: "memory");
}

// ---------------------------------------------------------------------------
// fp32 -> bf16 conversion pass
// ---------------------------------------------------------------------------
__global__ void cvt_bf16(const float* __restrict__ s, bf16_t* __restrict__ d, long long n) {
    long long i = (long long)blockIdx.x * blockDim.x + threadIdx.x;
    if (i < n) d[i] = tobf(s[i]);
}

// ---------------------------------------------------------------------------
// Fused GEMM v2: Y[b,m,n] = act(sum_k W[m,k]*X[b,k,n] + bias[m])
// W: [M,K] bf16, X: [B,K,HWD] bf16.  Block tile 128m x 128n, 8 waves
// (4m x 2n), wave tile 32x64 -> 8 WMMA / k-step.  A+B tiles staged by TDM,
// double buffered (issue next pair, s_wait_tensorcnt 2, compute, repeat).
// ACT: 0 none, 1 exact GELU, 2 SiLU.  WBF: also write bf16 copy of Y.
// ---------------------------------------------------------------------------
template <int ACT, bool WBF>
__global__ __launch_bounds__(256) void gemm_wmma(
    const bf16_t* __restrict__ W, const bf16_t* __restrict__ X,
    const float* __restrict__ bias, float* __restrict__ Y,
    bf16_t* __restrict__ Ybf, int M, int K)
{
    __shared__ __align__(16) bf16_t As[2 * 128 * 32];
    __shared__ __align__(16) bf16_t Bs[2 * 32 * 128];

    const int tid  = threadIdx.x;
    const int lane = tid & 31;
    const int wid  = tid >> 5;
    const int wm   = wid & 3;     // wave m-subtile 0..3 (32 rows each)
    const int wn   = wid >> 2;    // wave n-subtile 0..1 (64 cols each)
    const int half = lane >> 4;
    const int lr   = lane & 15;

    const int m0 = blockIdx.y * 128;
    const int n0 = blockIdx.x * 128;
    const size_t bofX = (size_t)blockIdx.z * (size_t)K * HWD;
    const size_t bofY = (size_t)blockIdx.z * (size_t)M * HWD;

    v8f acc[2][4];
#pragma unroll
    for (int mi = 0; mi < 2; ++mi)
#pragma unroll
        for (int nt = 0; nt < 4; ++nt) { v8f z = {}; acc[mi][nt] = z; }

    const int KT = K / 32;
    // prologue: stage k-tile 0 into buffer 0 (wave 0 drives the TDM)
    if (tid < 32) {
        tdm_load_2d(lds_off(&As[0]), W + (size_t)m0 * K, 32u, 128u, (unsigned long long)K);
        tdm_load_2d(lds_off(&Bs[0]), X + bofX + n0, 128u, 32u, (unsigned long long)HWD);
    }
    for (int kt = 0; kt < KT; ++kt) {
        const int buf = kt & 1;
        if (kt + 1 < KT) {
            const int nb = buf ^ 1;
            if (tid < 32) {
                tdm_load_2d(lds_off(&As[nb * 128 * 32]),
                            W + (size_t)m0 * K + (kt + 1) * 32, 32u, 128u,
                            (unsigned long long)K);
                tdm_load_2d(lds_off(&Bs[nb * 32 * 128]),
                            X + bofX + (size_t)((kt + 1) * 32) * HWD + n0, 128u, 32u,
                            (unsigned long long)HWD);
                __builtin_amdgcn_s_wait_tensorcnt(2);   // current pair done
            }
        } else if (tid < 32) {
            __builtin_amdgcn_s_wait_tensorcnt(0);
        }
        __syncthreads();

        const bf16_t* Ab = &As[buf * 128 * 32];
        const bf16_t* Bb = &Bs[buf * 32 * 128];
        v16bf a[2];
#pragma unroll
        for (int mi = 0; mi < 2; ++mi)
#pragma unroll
            for (int e = 0; e < 16; ++e)
                a[mi][e] = Ab[(wm * 32 + mi * 16 + lr) * 32 + ka_idx(e, half)];
#pragma unroll
        for (int nt = 0; nt < 4; ++nt) {
            v16bf bb;
#pragma unroll
            for (int e = 0; e < 16; ++e)
                bb[e] = Bb[kb_idx(e, half) * 128 + wn * 64 + nt * 16 + lr];
#pragma unroll
            for (int mi = 0; mi < 2; ++mi)
                acc[mi][nt] = __builtin_amdgcn_wmma_f32_16x16x32_bf16(
                    false, a[mi], false, bb, (short)0, acc[mi][nt], false, false);
        }
        __syncthreads();
    }

#pragma unroll
    for (int mi = 0; mi < 2; ++mi)
#pragma unroll
        for (int nt = 0; nt < 4; ++nt)
#pragma unroll
            for (int r = 0; r < 8; ++r) {
                int m = m0 + wm * 32 + mi * 16 + r + 8 * half;
                int n = n0 + wn * 64 + nt * 16 + lr;
                float y = acc[mi][nt][r];
                if (bias) y += bias[m];
                if (ACT == 1) y = 0.5f * y * (1.f + erff(y * 0.70710678118f));
                else if (ACT == 2) y = y / (1.f + __expf(-y));
                size_t o = bofY + (size_t)m * HWD + n;
                Y[o] = y;
                if (WBF) Ybf[o] = tobf(y);
            }
}

// ---------------------------------------------------------------------------
// Depthwise 3x3, SAME padding; writes fp32 (for norms) + bf16 (for WMMA).
// ---------------------------------------------------------------------------
__global__ void dwconv3x3(const float* __restrict__ X,
                          const float* __restrict__ Wd,
                          float* __restrict__ Y, bf16_t* __restrict__ Ybf)
{
    long long idx = (long long)blockIdx.x * blockDim.x + threadIdx.x;
    const long long total = (long long)BB * QKVC * HWD;
    if (idx >= total) return;
    int xx0 = (int)(idx & (WW - 1));
    int yy0 = (int)((idx >> 7) & (HH - 1));
    long long bc = idx >> 14;
    int ch = (int)(bc % QKVC);
    const float* w  = Wd + ch * 9;
    const float* in = X + bc * (long long)HWD;
    float s = 0.f;
#pragma unroll
    for (int dy = -1; dy <= 1; ++dy) {
        int yy = yy0 + dy;
        if (yy < 0 || yy >= HH) continue;
#pragma unroll
        for (int dx = -1; dx <= 1; ++dx) {
            int xc = xx0 + dx;
            if (xc < 0 || xc >= WW) continue;
            s += in[yy * WW + xc] * w[(dy + 1) * 3 + (dx + 1)];
        }
    }
    Y[idx] = s;
    Ybf[idx] = tobf(s);
}

// ---------------------------------------------------------------------------
// Row L2 norms of q,k channels (channels 0..767 of qkv_dw) per batch.
// ---------------------------------------------------------------------------
__global__ __launch_bounds__(256) void rownorm(const float* __restrict__ QK,
                                               float* __restrict__ norms)
{
    int b = blockIdx.x / 768, ch = blockIdx.x % 768;
    const float* row = QK + ((size_t)b * QKVC + ch) * HWD;
    float s = 0.f;
    for (int i = threadIdx.x; i < HWD; i += 256) { float v = row[i]; s += v * v; }
    __shared__ float red[256];
    red[threadIdx.x] = s;
    __syncthreads();
    for (int off = 128; off > 0; off >>= 1) {
        if (threadIdx.x < off) red[threadIdx.x] += red[threadIdx.x + off];
        __syncthreads();
    }
    if (threadIdx.x == 0) norms[blockIdx.x] = sqrtf(red[0]);
}

// ---------------------------------------------------------------------------
// Gram S[bh,c,d] = sum_n q[c,n]*k[d,n]; one workgroup (9 waves) per (b,h),
// wave (mi,ni) owns a 16x16 tile, contraction over HWD in bf16 chunks of 32.
// ---------------------------------------------------------------------------
__global__ __launch_bounds__(288) void gram_qk(const bf16_t* __restrict__ QKVDW,
                                               float* __restrict__ S)
{
    int bh = blockIdx.x, b = bh >> 3, h = bh & 7;
    int lane = threadIdx.x & 31, wid = threadIdx.x >> 5;
    int mi = wid / 3, ni = wid % 3;
    int half = lane >> 4, lr = lane & 15;

    const bf16_t* q = QKVDW + ((size_t)b * QKVC + h * CHD) * HWD;
    const bf16_t* k = QKVDW + ((size_t)b * QKVC + DIMC + h * CHD) * HWD;
    const bf16_t* qrow = q + (size_t)(mi * 16 + lr) * HWD;
    const bf16_t* krow = k + (size_t)(ni * 16 + lr) * HWD;

    int kia[16];
#pragma unroll
    for (int e = 0; e < 16; ++e) kia[e] = ka_idx(e, half);

    v8f acc = {};
    for (int kb = 0; kb < HWD; kb += 32) {
        v16bf a, bfr;
#pragma unroll
        for (int e = 0; e < 16; ++e) a[e] = qrow[kb + kia[e]];
#pragma unroll
        for (int e = 0; e < 16; ++e) bfr[e] = krow[kb + kb_idx(e, half)];
        acc = __builtin_amdgcn_wmma_f32_16x16x32_bf16(
            false, a, false, bfr, (short)0, acc, false, false);
    }
#pragma unroll
    for (int r = 0; r < 8; ++r) {
        int c = mi * 16 + r + 8 * half;
        int d = ni * 16 + lr;
        S[((size_t)bh * CHD + c) * CHD + d] = acc[r];
    }
}

// ---------------------------------------------------------------------------
// Attention post-process (scale, top-24 softmax, top-12 softmax of probs,
// P = attns[0]*A1 + attns[1]*A2, stored bf16 padded [bh,48,64]).
// ---------------------------------------------------------------------------
__global__ __launch_bounds__(256) void attn_post(
    const float* __restrict__ S, const float* __restrict__ norms,
    const float* __restrict__ temperature, const float* __restrict__ attns,
    bf16_t* __restrict__ P)
{
    int tid = blockIdx.x * 256 + threadIdx.x;
    if (tid >= BB * HEADS * CHD) return;
    int c = tid % CHD, h = (tid / CHD) % HEADS, b = tid / (CHD * HEADS);
    int bh = b * HEADS + h;

    float vals[CHD], a1[CHD], pr[CHD];
    float nq = fmaxf(norms[b * 768 + h * CHD + c], 1e-12f);
    float tmp = temperature[h];
    const float* srow = S + ((size_t)bh * CHD + c) * CHD;
    for (int d = 0; d < CHD; ++d) {
        float nk = fmaxf(norms[b * 768 + DIMC + h * CHD + d], 1e-12f);
        vals[d] = srow[d] / (nq * nk) * tmp;
    }
    float sc0 = attns[0], sc1 = attns[1];
    {   // round 1: keep top 24 (lax.top_k tie-break: earliest index)
        bool keep[CHD];
        float mx = -1e30f;
        for (int i = 0; i < CHD; ++i) {
            int rank = 0;
            for (int j = 0; j < CHD; ++j)
                if (vals[j] > vals[i] || (vals[j] == vals[i] && j < i)) rank++;
            keep[i] = (rank < 24);
            if (keep[i]) mx = fmaxf(mx, vals[i]);
        }
        float z = 0.f;
        for (int i = 0; i < CHD; ++i) { a1[i] = keep[i] ? __expf(vals[i] - mx) : 0.f; z += a1[i]; }
        for (int i = 0; i < CHD; ++i) a1[i] /= z;
    }
    {   // round 2: top 12 of the probabilities, softmax again
        bool keep[CHD];
        float mx = -1e30f;
        for (int i = 0; i < CHD; ++i) {
            int rank = 0;
            for (int j = 0; j < CHD; ++j)
                if (a1[j] > a1[i] || (a1[j] == a1[i] && j < i)) rank++;
            keep[i] = (rank < 12);
            if (keep[i]) mx = fmaxf(mx, a1[i]);
        }
        float z = 0.f;
        for (int i = 0; i < CHD; ++i) { pr[i] = keep[i] ? __expf(a1[i] - mx) : 0.f; z += pr[i]; }
        for (int i = 0; i < CHD; ++i) pr[i] = sc0 * a1[i] + sc1 * (pr[i] / z);
    }
    bf16_t* prow = P + ((size_t)bh * CHD + c) * 64;
    for (int d = 0; d < CHD; ++d) prow[d] = tobf(pr[d]);
    for (int d = CHD; d < 64; ++d) prow[d] = tobf(0.f);
}

// ---------------------------------------------------------------------------
// out_bf[b, h*48+c, n] = sum_d P[bh,c,d] * v[bh,d,n].  v tile staged via
// per-lane async global->LDS (ASYNCcnt), pad rows zeroed in LDS.
// ---------------------------------------------------------------------------
__global__ __launch_bounds__(256) void pv_wmma(const bf16_t* __restrict__ P,
                                               const bf16_t* __restrict__ QKVDW,
                                               bf16_t* __restrict__ OUTbf)
{
    __shared__ __align__(16) bf16_t Vs[64 * 128];
    int bh = blockIdx.y, b = bh >> 3, h = bh & 7;
    int n0 = blockIdx.x * 128;
    int tid = threadIdx.x, lane = tid & 31, wid = tid >> 5;
    int half = lane >> 4, lr = lane & 15;

    const bf16_t* v = QKVDW + ((size_t)b * QKVC + 2 * DIMC + h * CHD) * HWD;
    // 48 real rows x 16 b128-chunks, 3 async ops per thread
    for (int cidx = tid; cidx < CHD * 16; cidx += 256) {
        int r = cidx >> 4, q16 = cidx & 15;
        async_ld_b128(lds_off(&Vs[r * 128 + q16 * 8]),
                      v + (size_t)r * HWD + n0 + q16 * 8);
    }
    for (int i = tid; i < 16 * 128; i += 256) Vs[CHD * 128 + (i >> 7) * 0 + i] = tobf(0.f);
    wait_async0();
    __syncthreads();

    const bf16_t* Pbh = P + (size_t)bh * CHD * 64;
#pragma unroll
    for (int mi = 0; mi < 3; ++mi) {
        v8f acc = {};
#pragma unroll
        for (int kc = 0; kc < 2; ++kc) {
            v16bf a, bfr;
#pragma unroll
            for (int e = 0; e < 16; ++e)
                a[e] = Pbh[(mi * 16 + lr) * 64 + kc * 32 + ka_idx(e, half)];
#pragma unroll
            for (int e = 0; e < 16; ++e)
                bfr[e] = Vs[(kc * 32 + kb_idx(e, half)) * 128 + wid * 16 + lr];
            acc = __builtin_amdgcn_wmma_f32_16x16x32_bf16(
                false, a, false, bfr, (short)0, acc, false, false);
        }
#pragma unroll
        for (int r = 0; r < 8; ++r) {
            int c = mi * 16 + r + 8 * half;
            OUTbf[((size_t)b * DIMC + h * CHD + c) * HWD + n0 + wid * 16 + lr] = tobf(acc[r]);
        }
    }
}

// gate[b,n] = sigmoid(sum_c w2[c] * G[b,c,n] + b2)
__global__ void gate_kernel(const float* __restrict__ G,
                            const float* __restrict__ w2,
                            const float* __restrict__ b2,
                            float* __restrict__ gate)
{
    int idx = blockIdx.x * blockDim.x + threadIdx.x;
    if (idx >= BB * HWD) return;
    int b = idx / HWD, n = idx % HWD;
    const float* g = G + (size_t)b * DIMC * HWD + n;
    float s = b2[0];
    for (int c = 0; c < DIMC; ++c) s += w2[c] * g[(size_t)c * HWD];
    gate[idx] = 1.f / (1.f + __expf(-s));
}

// GPbf[b,c,n] = bf16(gate[b,n] * gk[b,c,n])   (c in 0..63)
__global__ void gate_mul(const float* __restrict__ gate,
                         const float* __restrict__ gk,
                         bf16_t* __restrict__ GPbf)
{
    long long idx = (long long)blockIdx.x * blockDim.x + threadIdx.x;
    const long long total = (long long)BB * 64 * HWD;
    if (idx >= total) return;
    int n = (int)(idx % HWD);
    int b = (int)(idx / ((long long)64 * HWD));
    GPbf[idx] = tobf(gate[b * HWD + n] * gk[idx]);
}

// alpha[b,n] = sum_c (z[c]/||z||) * gk1[b,c,n]
__global__ void alpha_kernel(const float* __restrict__ z,
                             const float* __restrict__ gk1,
                             float* __restrict__ alpha)
{
    int idx = blockIdx.x * blockDim.x + threadIdx.x;
    if (idx >= BB * HWD) return;
    int b = idx / HWD, n = idx % HWD;
    float nrm = 0.f;
    for (int c = 0; c < 64; ++c) nrm += z[c] * z[c];
    nrm = fmaxf(sqrtf(nrm), 1e-12f);
    float s = 0.f;
    for (int c = 0; c < 64; ++c)
        s += (z[c] / nrm) * gk1[((size_t)b * 64 + c) * HWD + n];
    alpha[idx] = s;
}

// Y = out + (1-alpha)*p_chr + alpha*p_detg
__global__ void combine_kernel(const float* __restrict__ outp,
                               const float* __restrict__ pchr,
                               const float* __restrict__ pdetg,
                               const float* __restrict__ alpha,
                               float* __restrict__ Y)
{
    long long idx = (long long)blockIdx.x * blockDim.x + threadIdx.x;
    const long long total = (long long)BB * DIMC * HWD;
    if (idx >= total) return;
    int n = (int)(idx % HWD);
    int b = (int)(idx / ((long long)DIMC * HWD));
    float al = alpha[b * HWD + n];
    Y[idx] = outp[idx] + (1.f - al) * pchr[idx] + al * pdetg[idx];
}

// ---------------------------------------------------------------------------
extern "C" void kernel_launch(void* const* d_in, const int* in_sizes, int n_in,
                              void* d_out, int out_size, void* d_ws, size_t ws_size,
                              hipStream_t stream)
{
    const float* x       = (const float*)d_in[0];
    const float* gk0     = (const float*)d_in[1];
    const float* gk1     = (const float*)d_in[2];
    const float* w_qkv   = (const float*)d_in[3];
    const float* w_dw    = (const float*)d_in[4];
    const float* w_proj  = (const float*)d_in[5];
    const float* temperature = (const float*)d_in[6];
    const float* attns   = (const float*)d_in[7];
    const float* detg_z  = (const float*)d_in[8];
    const float* chr_w1  = (const float*)d_in[9];
    const float* chr_b1  = (const float*)d_in[10];
    const float* chr_w2  = (const float*)d_in[11];
    const float* chr_b2  = (const float*)d_in[12];
    const float* chr_wt  = (const float*)d_in[13];
    const float* chr_bt  = (const float*)d_in[14];
    const float* detg_w1 = (const float*)d_in[15];
    const float* detg_b1 = (const float*)d_in[16];
    const float* detg_w2 = (const float*)d_in[17];
    const float* detg_b2 = (const float*)d_in[18];
    const float* detg_wt = (const float*)d_in[19];
    const float* detg_bt = (const float*)d_in[20];

    char* ws = (char*)d_ws;
    size_t off = 0;
    auto carve = [&](size_t bytes) -> void* {
        void* p = ws + off;
        off = (off + bytes + 255) & ~(size_t)255;
        return p;
    };

    const size_t NDH = (size_t)BB * DIMC * HWD;      // 12.58M
    const size_t NQH = (size_t)BB * QKVC * HWD;      // 37.75M

    float*  qkv_lin   = (float*)carve(NQH * 4);
    float*  qkv_dw    = (float*)carve(NQH * 4);
    bf16_t* qkv_dw_bf = (bf16_t*)carve(NQH * 2);
    bf16_t* x_bf      = (bf16_t*)carve(NDH * 2);
    bf16_t* wqkv_bf   = (bf16_t*)carve((size_t)QKVC * DIMC * 2);
    bf16_t* wproj_bf  = (bf16_t*)carve((size_t)DIMC * DIMC * 2);
    bf16_t* w1c_bf    = (bf16_t*)carve((size_t)DIMC * DIMC * 2);
    bf16_t* w1d_bf    = (bf16_t*)carve((size_t)DIMC * DIMC * 2);
    bf16_t* wtc_bf    = (bf16_t*)carve((size_t)DIMC * 64 * 2);
    bf16_t* wtd_bf    = (bf16_t*)carve((size_t)DIMC * 64 * 2);
    float*  norms     = (float*)carve((size_t)BB * 768 * 4);
    float*  Smat      = (float*)carve((size_t)BB * HEADS * CHD * CHD * 4);
    bf16_t* Pmat      = (bf16_t*)carve((size_t)BB * HEADS * CHD * 64 * 2);
    bf16_t* attn_bf   = (bf16_t*)carve(NDH * 2);
    float*  proj_out  = (float*)carve(NDH * 4);
    bf16_t* proj_bf   = (bf16_t*)carve(NDH * 2);
    float*  gbuf      = (float*)carve(NDH * 4);
    float*  gatebuf   = (float*)carve((size_t)BB * HWD * 4);
    bf16_t* gp_bf     = (bf16_t*)carve((size_t)BB * 64 * HWD * 2);
    float*  p_chr     = (float*)carve(NDH * 4);
    float*  p_detg    = (float*)carve(NDH * 4);
    float*  alphab    = (float*)carve((size_t)BB * HWD * 4);

    // 0. bf16 conversions (inputs + weights feeding WMMA GEMMs)
    cvt_bf16<<<(unsigned)((NDH + 255) / 256), 256, 0, stream>>>(x, x_bf, (long long)NDH);
    cvt_bf16<<<(QKVC * DIMC) / 256, 256, 0, stream>>>(w_qkv, wqkv_bf, QKVC * DIMC);
    cvt_bf16<<<(DIMC * DIMC) / 256, 256, 0, stream>>>(w_proj, wproj_bf, DIMC * DIMC);
    cvt_bf16<<<(DIMC * DIMC) / 256, 256, 0, stream>>>(chr_w1, w1c_bf, DIMC * DIMC);
    cvt_bf16<<<(DIMC * DIMC) / 256, 256, 0, stream>>>(detg_w1, w1d_bf, DIMC * DIMC);
    cvt_bf16<<<(DIMC * 64) / 256, 256, 0, stream>>>(chr_wt, wtc_bf, DIMC * 64);
    cvt_bf16<<<(DIMC * 64) / 256, 256, 0, stream>>>(detg_wt, wtd_bf, DIMC * 64);

    // 1. qkv = w_qkv @ x
    gemm_wmma<0, false><<<dim3(HWD / 128, QKVC / 128, BB), 256, 0, stream>>>(
        wqkv_bf, x_bf, nullptr, qkv_lin, nullptr, QKVC, DIMC);
    // 2. depthwise 3x3 (fp32 + bf16 outputs)
    dwconv3x3<<<(unsigned)(NQH / 256), 256, 0, stream>>>(qkv_lin, w_dw, qkv_dw, qkv_dw_bf);
    // 3. q/k row norms
    rownorm<<<BB * 768, 256, 0, stream>>>(qkv_dw, norms);
    // 4. Gram q.kT
    gram_qk<<<BB * HEADS, 288, 0, stream>>>(qkv_dw_bf, Smat);
    // 5. top-k / softmax -> P
    attn_post<<<(BB * HEADS * CHD + 255) / 256, 256, 0, stream>>>(
        Smat, norms, temperature, attns, Pmat);
    // 6. out = P @ v  (bf16 out, feeds proj GEMM)
    pv_wmma<<<dim3(HWD / 128, BB * HEADS), 256, 0, stream>>>(Pmat, qkv_dw_bf, attn_bf);
    // 7. proj (fp32 for residual/combine + bf16 for the two prompt GEMMs)
    gemm_wmma<0, true><<<dim3(HWD / 128, DIMC / 128, BB), 256, 0, stream>>>(
        wproj_bf, attn_bf, nullptr, proj_out, proj_bf, DIMC, DIMC);

    // 8. chr prompt branch
    gemm_wmma<1, false><<<dim3(HWD / 128, DIMC / 128, BB), 256, 0, stream>>>(
        w1c_bf, proj_bf, chr_b1, gbuf, nullptr, DIMC, DIMC);
    gate_kernel<<<(BB * HWD) / 256, 256, 0, stream>>>(gbuf, chr_w2, chr_b2, gatebuf);
    gate_mul<<<(BB * 64 * HWD) / 256, 256, 0, stream>>>(gatebuf, gk0, gp_bf);
    gemm_wmma<2, false><<<dim3(HWD / 128, DIMC / 128, BB), 256, 0, stream>>>(
        wtc_bf, gp_bf, chr_bt, p_chr, nullptr, DIMC, 64);

    // 9. detg prompt branch
    gemm_wmma<1, false><<<dim3(HWD / 128, DIMC / 128, BB), 256, 0, stream>>>(
        w1d_bf, proj_bf, detg_b1, gbuf, nullptr, DIMC, DIMC);
    gate_kernel<<<(BB * HWD) / 256, 256, 0, stream>>>(gbuf, detg_w2, detg_b2, gatebuf);
    gate_mul<<<(BB * 64 * HWD) / 256, 256, 0, stream>>>(gatebuf, gk1, gp_bf);
    gemm_wmma<2, false><<<dim3(HWD / 128, DIMC / 128, BB), 256, 0, stream>>>(
        wtd_bf, gp_bf, detg_bt, p_detg, nullptr, DIMC, 64);

    // 10. alpha, 11. combine
    alpha_kernel<<<(BB * HWD) / 256, 256, 0, stream>>>(detg_z, gk1, alphab);
    combine_kernel<<<(unsigned)(NDH / 256), 256, 0, stream>>>(
        proj_out, p_chr, p_detg, alphab, (float*)d_out);
}
